// Swin_Frame_Attention_4260607558002
// MI455X (gfx1250) — matrix operand order repaired and checked by reading, not compiled, not run on hardware
//
#include <hip/hip_runtime.h>

// ---------------------------------------------------------------------------
// Swin frame cross-attention for MI455X (gfx1250, wave32, WMMA bf16 path).
//
// B=2, C=384, T=2, H=W=192, WIN=12 -> S=144/window, nW=256, 8 heads, hd=48.
// ~2.5e11 FLOPs vs ~0.5 GB mandatory HBM traffic (21 us @ 23.3 TB/s): the ten
// 73728x384x384 GEMMs + windowed attention run on v_wmma_f32_16x16x32_bf16
// (f32 accumulate); intermediates are bf16 (57 MB each -> L2-resident).
//
// Round-4: fix async builtin pointer types (takes v4i32 AS1 src / AS3 dst).
// Keeps: guarded v_cvt_pk_bf16_f32, GLOBAL_LOAD_ASYNC_TO_LDS_B128 +
// s_wait_asynccnt for the bf16 A-tiles, double-buffered K-loops, packed LDS
// fills, per-block scatter tables, attention LDS ~145 KB (2 blocks/WGP).
// ---------------------------------------------------------------------------

typedef __bf16 bf16;
typedef __attribute__((ext_vector_type(16))) __bf16 v16bf;
typedef __attribute__((ext_vector_type(8)))  __bf16 v8bf;
typedef __attribute__((ext_vector_type(2)))  __bf16 v2bf;
typedef __attribute__((ext_vector_type(8)))  float  v8f;
typedef int v4i_gcc __attribute__((vector_size(16)));   // matches builtin param

#define AS1 __attribute__((address_space(1)))
#define AS3 __attribute__((address_space(3)))

#if defined(__has_builtin)
#  if __has_builtin(__builtin_amdgcn_cvt_pk_bf16_f32)
#    define HAVE_CVT_PK_BF16 1
#  endif
#  if __has_builtin(__builtin_amdgcn_global_load_async_to_lds_b128)
#    define USE_ASYNC_LDS 1
#  endif
#endif
#ifndef HAVE_CVT_PK_BF16
#  define HAVE_CVT_PK_BF16 0
#endif
#ifndef USE_ASYNC_LDS
#  define USE_ASYNC_LDS 0
#endif

#define HEADS   8
#define HD      48
#define WINSZ   12
#define S_WIN   144
#define NW      256
#define C_DIM   384
#define IMG     192
#define HW_IMG  (IMG * IMG)      // 36864 (divisible by BM)
#define BATCH   2
#define M_TOT   (BATCH * HW_IMG) // 73728

#define BM 128
#define BN 64
#define BK 32
#define KW (BK / 2)              // packed u32 words per row
#define SPAD 160                 // S_WIN padded to multiple of 32
#define PLD (2 * SPAD)           // bf16 ld of P aliased onto f32 S rows

__device__ __forceinline__ unsigned bfbits(float f) {
  union { float f; unsigned u; } v; v.f = f;
  unsigned r = v.u + 0x7FFFu + ((v.u >> 16) & 1u);   // RNE
  return r >> 16;
}
__device__ __forceinline__ unsigned pack2bf(float a, float b) {
#if HAVE_CVT_PK_BF16
  v2bf p = __builtin_amdgcn_cvt_pk_bf16_f32(a, b);
  return __builtin_bit_cast(unsigned, p);
#else
  return (bfbits(b) << 16) | bfbits(a);
#endif
}
__device__ __forceinline__ bf16 f2bf(float f) {
  unsigned short h = (unsigned short)bfbits(f);
  return __builtin_bit_cast(bf16, h);
}
__device__ __forceinline__ void wait_async0() {
#if defined(__has_builtin) && __has_builtin(__builtin_amdgcn_s_wait_asynccnt)
  __builtin_amdgcn_s_wait_asynccnt(0);
#else
  asm volatile("s_wait_asynccnt 0x0" ::: "memory");
#endif
}

// A fragment (16x32 bf16, rows striped over lanes): two contiguous 16B reads.
__device__ __forceinline__ v16bf load_frag_a(const bf16* base, int ld) {
  const int lane = threadIdx.x & 31;
  const int row  = lane & 15;
  const int kb   = (lane >> 4) << 3;          // 0 or 8
  const bf16* p = base + row * ld + kb;
  v8bf lo = *(const v8bf*)(p);
  v8bf hi = *(const v8bf*)(p + 16);
  v16bf f;
#pragma unroll
  for (int i = 0; i < 8; ++i) { f[i] = lo[i]; f[i + 8] = hi[i]; }
  return f;
}
// B fragment (32x16 bf16, cols striped): K-contiguous tile bt[col][k], two 16B reads.
__device__ __forceinline__ v16bf load_frag_b(const bf16* bt, int ld) {
  const int lane = threadIdx.x & 31;
  const int col  = lane & 15;
  const int kb   = (lane >> 4) << 3;          // 0 or 8
  const bf16* p = bt + col * ld + kb;
  v8bf lo = *(const v8bf*)(p);
  v8bf hi = *(const v8bf*)(p + 16);
  v16bf f;
#pragma unroll
  for (int i = 0; i < 8; ++i) { f[i] = lo[i]; f[i + 8] = hi[i]; }
  return f;
}
__device__ __forceinline__ v8f wmma_bf16(v16bf a, v16bf b, v8f c) {
  return __builtin_amdgcn_wmma_f32_16x16x32_bf16(false, a, false, b,
                                                 (short)0, c, false, false);
}

// ---------------------------------------------------------------------------
// Kernel 1: Y[(b,wi,s), n] = sum_c x[b,c,frame,h,w] * W[c,n]   (window-scattered)
// ---------------------------------------------------------------------------
__global__ __launch_bounds__(256)
void qkv_gemm_kernel(const float* __restrict__ x, const float* __restrict__ W,
                     bf16* __restrict__ out, int frame) {
  __shared__ __align__(64) unsigned AsU[2][BM * KW];  // [m][k/2] packed bf16x2
  __shared__ __align__(64) unsigned BsU[2][BN * KW];  // [n][k/2] (K-contiguous)
  __shared__ unsigned outIdx[BM];

  const int tid = threadIdx.x;
  const int m0  = blockIdx.x * BM;
  const int n0  = blockIdx.y * BN;
  const int b   = m0 / HW_IMG;                 // block-uniform (BM | HW_IMG)
  const int hw0 = m0 - b * HW_IMG;
  const size_t xbase = ((size_t)(b * C_DIM) * 2 + frame) * HW_IMG + hw0;

  if (tid < BM) {                              // scatter table, once per block
    int hw = hw0 + tid;
    int h = hw / IMG, w = hw - h * IMG;
    int wi = (h / WINSZ) * 16 + (w / WINSZ);
    int s  = (h % WINSZ) * WINSZ + (w % WINSZ);
    outIdx[tid] = (unsigned)((b * NW + wi) * S_WIN + s) * C_DIM;
  }

  const int wave = tid >> 5;
  const int wm = wave & 3, wn = wave >> 2;
  v8f acc[2][2];
#pragma unroll
  for (int i = 0; i < 2; ++i)
#pragma unroll
    for (int j = 0; j < 2; ++j) acc[i][j] = (v8f){};

  unsigned rA[8], rB[4];
  auto loadAB = [&](int k0) {
#pragma unroll
    for (int j = 0; j < 8; ++j) {              // A: 8 packed words / thread
      int p = tid + j * 256;
      int ml = p & (BM - 1), kp = p >> 7;
      int k = k0 + 2 * kp;
      float a0 = x[xbase + (size_t)k * (2 * HW_IMG) + ml];
      float a1 = x[xbase + (size_t)(k + 1) * (2 * HW_IMG) + ml];
      rA[j] = pack2bf(a0, a1);
    }
#pragma unroll
    for (int j = 0; j < 4; ++j) {              // B: 4 packed words / thread
      int p = tid + j * 256;
      int nn = p & (BN - 1), kp = p >> 6;
      int k = k0 + 2 * kp;
      float b0 = W[(size_t)k * C_DIM + n0 + nn];
      float b1 = W[(size_t)(k + 1) * C_DIM + n0 + nn];
      rB[j] = pack2bf(b0, b1);
    }
  };
  auto storeAB = [&](int buf) {
#pragma unroll
    for (int j = 0; j < 8; ++j) {
      int p = tid + j * 256;
      AsU[buf][(p & (BM - 1)) * KW + (p >> 7)] = rA[j];
    }
#pragma unroll
    for (int j = 0; j < 4; ++j) {
      int p = tid + j * 256;
      BsU[buf][(p & (BN - 1)) * KW + (p >> 6)] = rB[j];
    }
  };

  loadAB(0);
  storeAB(0);
  __syncthreads();

  int buf = 0;
  for (int k0 = 0; k0 < C_DIM; k0 += BK, buf ^= 1) {
    const bool more = (k0 + BK) < C_DIM;
    if (more) loadAB(k0 + BK);                 // prefetch next slab
    const bf16* As = (const bf16*)AsU[buf];
    const bf16* Bs = (const bf16*)BsU[buf];
#pragma unroll
    for (int i = 0; i < 2; ++i) {
      v16bf a = load_frag_a(As + (wm * 32 + i * 16) * BK, BK);
#pragma unroll
      for (int j = 0; j < 2; ++j) {
        v16bf bb = load_frag_b(Bs + (wn * 32 + j * 16) * BK, BK);
        acc[i][j] = wmma_bf16(a, bb, acc[i][j]);
      }
    }
    if (more) storeAB(buf ^ 1);
    __syncthreads();
  }

  const int lane = tid & 31, colL = lane & 15, rb = (lane >> 4) << 3;
#pragma unroll
  for (int i = 0; i < 2; ++i)
#pragma unroll
    for (int j = 0; j < 2; ++j) {
      int n = n0 + wn * 32 + j * 16 + colL;
#pragma unroll
      for (int r = 0; r < 8; ++r) {
        int ml = wm * 32 + i * 16 + rb + r;
        out[(size_t)outIdx[ml] + n] = f2bf(acc[i][j][r]);
      }
    }
}

// ---------------------------------------------------------------------------
// Kernel 2: windowed attention, one block per (window, head, batch).
// LDS: Q[144x64] K[144x64] (48->64 pad), Vt[48x160] (transposed, s-pad),
//      S[144x160] f32 (P bf16 aliased on top), bias[144]  => 144,960 B.
// ---------------------------------------------------------------------------
__global__ __launch_bounds__(256)
void attn_kernel(const bf16* __restrict__ Q, const bf16* __restrict__ K,
                 const bf16* __restrict__ V, const int* __restrict__ mask,
                 bf16* __restrict__ O) {
  extern __shared__ v8f dyn_smem[];
  char*  raw   = (char*)dyn_smem;
  bf16*  Qs    = (bf16*)raw;                   // 144*64   18432 B
  bf16*  Ks    = Qs + S_WIN * 64;              // 144*64   18432 B
  bf16*  Vt    = Ks + S_WIN * 64;              // 48*160   15360 B
  float* Sm    = (float*)(Vt + HD * SPAD);     // 144*160  92160 B
  float* biasS = Sm + S_WIN * SPAD;            // 144        576 B
  bf16*  Ps    = (bf16*)Sm;                    // aliased, row stride PLD=320

  const int wi = blockIdx.x, head = blockIdx.y, b = blockIdx.z;
  const int tid = threadIdx.x, wave = tid >> 5;
  const size_t base = ((size_t)(b * NW + wi)) * S_WIN * C_DIM + head * HD;

  // Q/K fill: packed u32 (two bf16), 4608 words each.
  for (int idx = tid; idx < S_WIN * 32; idx += 256) {
    int s = idx >> 5, cw = idx & 31;
    unsigned w = (cw < HD / 2)
        ? *(const unsigned*)(Q + base + (size_t)s * C_DIM + 2 * cw) : 0u;
    ((unsigned*)Qs)[s * 32 + cw] = w;
    unsigned w2 = (cw < HD / 2)
        ? *(const unsigned*)(K + base + (size_t)s * C_DIM + 2 * cw) : 0u;
    ((unsigned*)Ks)[s * 32 + cw] = w2;
  }
  // Vt fill (transposed, gather): packed pairs along s.
  for (int idx = tid; idx < HD * (SPAD / 2); idx += 256) {
    int c = idx / (SPAD / 2), sw = idx - c * (SPAD / 2);
    int s = 2 * sw;
    unsigned lo = (s < S_WIN)
        ? (unsigned)__builtin_bit_cast(unsigned short, V[base + (size_t)s * C_DIM + c]) : 0u;
    unsigned hi = (s + 1 < S_WIN)
        ? (unsigned)__builtin_bit_cast(unsigned short, V[base + (size_t)(s + 1) * C_DIM + c]) : 0u;
    ((unsigned*)Vt)[idx] = (hi << 16) | lo;
  }
  for (int s = tid; s < S_WIN; s += 256) {     // bias from mask[:,1,0]
    int h = (wi >> 4) * WINSZ + s / WINSZ;
    int w = (wi & 15) * WINSZ + s % WINSZ;
    int mv = mask[((size_t)b * 2 + 1) * HW_IMG + h * IMG + w];
    biasS[s] = (mv == 0) ? -1.0e9f : 0.0f;
  }
  __syncthreads();

  // S = Q * K^T : 9x9 tiles, K-dim 64 (two WMMA steps).
  for (int t = wave; t < 81; t += 8) {
    int tm = t / 9, tn = t % 9;
    v8f acc = (v8f){};
#pragma unroll
    for (int k0 = 0; k0 < 64; k0 += 32) {
      v16bf a  = load_frag_a(Qs + tm * 16 * 64 + k0, 64);
      v16bf bb = load_frag_b(Ks + tn * 16 * 64 + k0, 64);
      acc = wmma_bf16(a, bb, acc);
    }
    int lane = tid & 31, col = lane & 15, rb = (lane >> 4) << 3;
#pragma unroll
    for (int r = 0; r < 8; ++r)
      Sm[(tm * 16 + rb + r) * SPAD + tn * 16 + col] = acc[r];
  }
  __syncthreads();

  // Row softmax; P (bf16, packed u32 writes) aliases the S buffer. Each word
  // write lands at float slot k <= first read slot 2k, so no self-clobber.
  const float scale = 0.14433756729740643f;    // 1/sqrt(48)
  if (tid < S_WIN) {
    float* row = Sm + tid * SPAD;
    float mx = -3.0e38f;
    for (int k = 0; k < S_WIN; ++k) {
      float v = row[k] * scale + biasS[k];
      row[k] = v;
      mx = fmaxf(mx, v);
    }
    float sum = 0.0f;
    for (int k = 0; k < S_WIN; ++k) {
      float e = __expf(row[k] - mx);
      row[k] = e;
      sum += e;
    }
    float inv = 1.0f / sum;
    unsigned* prow = (unsigned*)row;
    for (int k = 0; k < S_WIN / 2; ++k)
      prow[k] = pack2bf(row[2 * k] * inv, row[2 * k + 1] * inv);
    for (int k = S_WIN / 2; k < SPAD / 2; ++k) prow[k] = 0u;   // K padding
  }
  __syncthreads();

  // O = P * V : 9x3 tiles, K-dim 160 (five WMMA steps). P ld = PLD (aliased).
  for (int t = wave; t < 27; t += 8) {
    int tm = t / 3, tn = t % 3;
    v8f acc = (v8f){};
#pragma unroll
    for (int k0 = 0; k0 < SPAD; k0 += 32) {
      v16bf a  = load_frag_a(Ps + tm * 16 * PLD + k0, PLD);
      v16bf bb = load_frag_b(Vt + tn * 16 * SPAD + k0, SPAD);
      acc = wmma_bf16(a, bb, acc);
    }
    int lane = tid & 31, col = lane & 15, rb = (lane >> 4) << 3;
#pragma unroll
    for (int r = 0; r < 8; ++r) {
      int srow = tm * 16 + rb + r;
      int c    = head * HD + tn * 16 + col;
      O[((size_t)(b * NW + wi) * S_WIN + srow) * C_DIM + c] = f2bf(acc[r]);
    }
  }
}

// ---------------------------------------------------------------------------
// Kernel 3: Y[m,n] = sum_k A[m,k]*W(k,n) + bias[n].
//   A: window-ordered bf16 (async-copied straight into LDS when available).
//   transposeW=1 -> W[n][k] ('dc').  outBf: bf16.  outF32: scatter to d_out.
// ---------------------------------------------------------------------------
__global__ __launch_bounds__(256)
void rowmajor_gemm_kernel(const bf16* __restrict__ A, const float* __restrict__ W,
                          const float* __restrict__ bias,
                          bf16* __restrict__ outBf, float* __restrict__ outF32,
                          int transposeW, int frame) {
  __shared__ __align__(64) unsigned AsU[2][BM * KW];
  __shared__ __align__(64) unsigned BsU[2][BN * KW];
  __shared__ unsigned outIdx[BM];

  const int tid = threadIdx.x;
  const int m0  = blockIdx.x * BM;
  const int n0  = blockIdx.y * BN;
  const int b   = m0 / (NW * S_WIN);           // block-uniform

  if (tid < BM) {
    if (outF32) {
      int rem = (m0 - b * (NW * S_WIN)) + tid;
      int wi = rem / S_WIN, s = rem - wi * S_WIN;
      int h = (wi >> 4) * WINSZ + s / WINSZ;
      int w = (wi & 15) * WINSZ + s % WINSZ;
      outIdx[tid] = (unsigned)((b * C_DIM * 2 + frame) * HW_IMG + h * IMG + w);
    } else {
      outIdx[tid] = (unsigned)(m0 + tid) * C_DIM;
    }
  }

  const int wave = tid >> 5;
  const int wm = wave & 3, wn = wave >> 2;
  v8f acc[2][2];
#pragma unroll
  for (int i = 0; i < 2; ++i)
#pragma unroll
    for (int j = 0; j < 2; ++j) acc[i][j] = (v8f){};

  unsigned rB[4];
#if !USE_ASYNC_LDS
  uint4 rA[2];
#endif

  auto loadB = [&](int k0) {
#pragma unroll
    for (int j = 0; j < 4; ++j) {
      int p = tid + j * 256;
      if (transposeW) {
        int kp = p & (KW - 1), nn = p >> 4;
        const float* wp = W + (size_t)(n0 + nn) * C_DIM + k0 + 2 * kp;
        rB[j] = pack2bf(wp[0], wp[1]);
      } else {
        int nn = p & (BN - 1), kp = p >> 6;
        int k = k0 + 2 * kp;
        rB[j] = pack2bf(W[(size_t)k * C_DIM + n0 + nn],
                        W[(size_t)(k + 1) * C_DIM + n0 + nn]);
      }
    }
  };
  auto storeB = [&](int buf) {
#pragma unroll
    for (int j = 0; j < 4; ++j) {
      int p = tid + j * 256;
      if (transposeW) BsU[buf][(p >> 4) * KW + (p & (KW - 1))] = rB[j];
      else            BsU[buf][(p & (BN - 1)) * KW + (p >> 6)] = rB[j];
    }
  };
#if USE_ASYNC_LDS
  // bf16 A-tile: DMA 16B chunks straight into LDS (ASYNCcnt-tracked).
  auto asyncA = [&](int k0, int buf) {
#pragma unroll
    for (int j = 0; j < 2; ++j) {
      int p = tid + j * 256;
      int ml = p >> 2, q = p & 3;
      const bf16* gp = A + (size_t)(m0 + ml) * C_DIM + k0 + 8 * q;
      unsigned* lp = &AsU[buf][ml * KW + 4 * q];
      __builtin_amdgcn_global_load_async_to_lds_b128(
          (AS1 v4i_gcc*)gp, (AS3 v4i_gcc*)lp, 0, 0);
    }
  };
#else
  auto loadA = [&](int k0) {
#pragma unroll
    for (int j = 0; j < 2; ++j) {
      int p = tid + j * 256;
      int ml = p >> 2, q = p & 3;
      rA[j] = *(const uint4*)(A + (size_t)(m0 + ml) * C_DIM + k0 + 8 * q);
    }
  };
  auto storeA = [&](int buf) {
#pragma unroll
    for (int j = 0; j < 2; ++j) {
      int p = tid + j * 256;
      int ml = p >> 2, q = p & 3;
      *(uint4*)&AsU[buf][ml * KW + 4 * q] = rA[j];
    }
  };
#endif

#if USE_ASYNC_LDS
  asyncA(0, 0);
  loadB(0);
  storeB(0);
  wait_async0();
#else
  loadA(0);
  loadB(0);
  storeA(0);
  storeB(0);
#endif
  __syncthreads();

  int buf = 0;
  for (int k0 = 0; k0 < C_DIM; k0 += BK, buf ^= 1) {
    const bool more = (k0 + BK) < C_DIM;
    if (more) {
#if USE_ASYNC_LDS
      asyncA(k0 + BK, buf ^ 1);
#else
      loadA(k0 + BK);
#endif
      loadB(k0 + BK);
    }
    const bf16* As = (const bf16*)AsU[buf];
    const bf16* Bs = (const bf16*)BsU[buf];
#pragma unroll
    for (int i = 0; i < 2; ++i) {
      v16bf a = load_frag_a(As + (wm * 32 + i * 16) * BK, BK);
#pragma unroll
      for (int j = 0; j < 2; ++j) {
        v16bf bb = load_frag_b(Bs + (wn * 32 + j * 16) * BK, BK);
        acc[i][j] = wmma_bf16(a, bb, acc[i][j]);
      }
    }
    if (more) {
#if !USE_ASYNC_LDS
      storeA(buf ^ 1);
#endif
      storeB(buf ^ 1);
    }
#if USE_ASYNC_LDS
    wait_async0();
#endif
    __syncthreads();
  }

  const int lane = tid & 31, colL = lane & 15, rb = (lane >> 4) << 3;
#pragma unroll
  for (int i = 0; i < 2; ++i)
#pragma unroll
    for (int j = 0; j < 2; ++j) {
      int n = n0 + wn * 32 + j * 16 + colL;
      float bv = bias[n];
#pragma unroll
      for (int r = 0; r < 8; ++r) {
        int ml = wm * 32 + i * 16 + rb + r;
        float v = acc[i][j][r] + bv;
        if (outF32) outF32[(size_t)outIdx[ml] + (size_t)n * (2 * HW_IMG)] = v;
        else        outBf[(size_t)outIdx[ml] + n] = f2bf(v);
      }
    }
}

// ---------------------------------------------------------------------------
// Launch. Workspace: 5 bf16 buffers of M_TOT*C_DIM (57 MB each, ~283 MB).
// ---------------------------------------------------------------------------
extern "C" void kernel_launch(void* const* d_in, const int* in_sizes, int n_in,
                              void* d_out, int out_size, void* d_ws, size_t ws_size,
                              hipStream_t stream) {
  (void)in_sizes; (void)n_in; (void)out_size; (void)ws_size;
  const float* x    = (const float*)d_in[0];
  const int*   mask = (const int*)d_in[1];
  const float* wq1  = (const float*)d_in[2];
  const float* wk1  = (const float*)d_in[3];
  const float* wv1  = (const float*)d_in[4];
  const float* wo1  = (const float*)d_in[5];
  const float* wq2  = (const float*)d_in[6];
  const float* wk2  = (const float*)d_in[7];
  const float* wv2  = (const float*)d_in[8];
  const float* wo2  = (const float*)d_in[9];
  const float* pw1  = (const float*)d_in[10];
  const float* pw2  = (const float*)d_in[11];
  const float* bo1  = (const float*)d_in[12];
  const float* bo2  = (const float*)d_in[13];
  const float* pb1  = (const float*)d_in[14];
  const float* pb2  = (const float*)d_in[15];
  float* out = (float*)d_out;

  const size_t elems = (size_t)M_TOT * C_DIM;
  bf16* Qb = (bf16*)d_ws;
  bf16* Kb = Qb + elems;
  bf16* Vb = Kb + elems;
  bf16* Ob = Vb + elems;
  bf16* Pb = Ob + elems;

  const int ATTN_SMEM = S_WIN * 64 * 2 * 2      // Qs + Ks
                      + HD * SPAD * 2           // Vt
                      + S_WIN * SPAD * 4        // Sm (P aliased)
                      + S_WIN * 4;              // bias   => 144,960 B (2/WGP)
  (void)hipFuncSetAttribute(reinterpret_cast<const void*>(attn_kernel),
                            hipFuncAttributeMaxDynamicSharedMemorySize, ATTN_SMEM);

  dim3 blk(256);
  dim3 gGrid(M_TOT / BM, C_DIM / BN);   // (576, 6)
  dim3 aGrid(NW, HEADS, BATCH);         // (256, 8, 2)

  // ---- SAM1: q from frame0, k/v from frame1, output -> t=0 ----
  qkv_gemm_kernel<<<gGrid, blk, 0, stream>>>(x, wq1, Qb, 0);
  qkv_gemm_kernel<<<gGrid, blk, 0, stream>>>(x, wk1, Kb, 1);
  qkv_gemm_kernel<<<gGrid, blk, 0, stream>>>(x, wv1, Vb, 1);
  attn_kernel<<<aGrid, blk, ATTN_SMEM, stream>>>(Qb, Kb, Vb, mask, Ob);
  rowmajor_gemm_kernel<<<gGrid, blk, 0, stream>>>(Ob, wo1, bo1, Pb, nullptr, 0, 0);
  rowmajor_gemm_kernel<<<gGrid, blk, 0, stream>>>(Pb, pw1, pb1, nullptr, out, 1, 0);

  // ---- SAM2: q from frame1, k/v from frame0, output -> t=1 ----
  qkv_gemm_kernel<<<gGrid, blk, 0, stream>>>(x, wq2, Qb, 1);
  qkv_gemm_kernel<<<gGrid, blk, 0, stream>>>(x, wk2, Kb, 0);
  qkv_gemm_kernel<<<gGrid, blk, 0, stream>>>(x, wv2, Vb, 0);
  attn_kernel<<<aGrid, blk, ATTN_SMEM, stream>>>(Qb, Kb, Vb, mask, Ob);
  rowmajor_gemm_kernel<<<gGrid, blk, 0, stream>>>(Ob, wo2, bo2, Pb, nullptr, 0, 0);
  rowmajor_gemm_kernel<<<gGrid, blk, 0, stream>>>(Pb, pw2, pb2, nullptr, out, 1, 1);
}
